// MultiHeadAttention_4458176053264
// MI455X (gfx1250) — matrix operand order repaired and checked by reading, not compiled, not run on hardware
//
#include <hip/hip_runtime.h>
#include <cstdint>

// ---------------------------------------------------------------------------
// MI455X (gfx1250) multi-head causal attention, bf16 WMMA + flash softmax.
// ---------------------------------------------------------------------------

typedef __attribute__((ext_vector_type(16))) __bf16 v16bf;
typedef __attribute__((ext_vector_type(8)))  __bf16 v8bf;
typedef __attribute__((ext_vector_type(8)))  float  v8f;

#define DMODEL 1024
#define SEQ    2048
#define NBATCH 2
#define NHEAD  16
#define DHEAD  64
#define LDS_STRIDE 40   // 32 + 8 pad, keeps 16B alignment (40*2 = 80 bytes)

static __device__ __forceinline__ v16bf frag_cat(v8bf lo, v8bf hi) {
  return __builtin_shufflevector(lo, hi, 0,1,2,3,4,5,6,7,8,9,10,11,12,13,14,15);
}
// A-fragment (16x32, M-rows per lane). p = &row[k0 + half*16]; 16 contiguous.
static __device__ __forceinline__ v16bf ldA(const __bf16* p) {
  return frag_cat(*(const v8bf*)p, *(const v8bf*)(p + 8));
}
// B-fragment (32x16, N-cols per lane). p = &col[k0 + half*16]; 16 contiguous.
static __device__ __forceinline__ v16bf ldB(const __bf16* p) {
  return frag_cat(*(const v8bf*)p, *(const v8bf*)(p + 8));
}
static __device__ __forceinline__ v8f wmma_bf16(v16bf a, v16bf b, v8f c) {
  return __builtin_amdgcn_wmma_f32_16x16x32_bf16(false, a, false, b,
                                                 (short)0, c, false, false);
}
static __device__ __forceinline__ float redmax16(float v) {
  v = fmaxf(v, __shfl_xor(v, 1, 32));
  v = fmaxf(v, __shfl_xor(v, 2, 32));
  v = fmaxf(v, __shfl_xor(v, 4, 32));
  v = fmaxf(v, __shfl_xor(v, 8, 32));
  return v;
}
static __device__ __forceinline__ float redsum16(float v) {
  v += __shfl_xor(v, 1, 32);
  v += __shfl_xor(v, 2, 32);
  v += __shfl_xor(v, 4, 32);
  v += __shfl_xor(v, 8, 32);
  return v;
}

// ---------------------------------------------------------------------------
// Kernel 1: fp32 -> bf16 elementwise (for x). 8 elems/thread, b128 stores.
// ---------------------------------------------------------------------------
__global__ __launch_bounds__(256) void cvt_f32_to_bf16(
    const float* __restrict__ in, __bf16* __restrict__ out, int n) {
  int i = (blockIdx.x * blockDim.x + threadIdx.x) * 8;
  if (i + 7 >= n) return;
  float4 a = *(const float4*)(in + i);
  float4 b = *(const float4*)(in + i + 4);
  v8bf o;
  o[0] = (__bf16)a.x; o[1] = (__bf16)a.y; o[2] = (__bf16)a.z; o[3] = (__bf16)a.w;
  o[4] = (__bf16)b.x; o[5] = (__bf16)b.y; o[6] = (__bf16)b.z; o[7] = (__bf16)b.w;
  *(v8bf*)(out + i) = o;
}

// ---------------------------------------------------------------------------
// Kernel 2: transpose + convert W (fp32 [k][n]) -> Wt (bf16 [n][k]).
// blockDim (32,8); 32x32 tile through padded LDS. z selects Wq/Wk/Wv.
// ---------------------------------------------------------------------------
__global__ __launch_bounds__(256) void transpose_cvt_w(
    const float* __restrict__ Wq, const float* __restrict__ Wk,
    const float* __restrict__ Wv, __bf16* __restrict__ Wqt,
    __bf16* __restrict__ Wkt, __bf16* __restrict__ Wvt) {
  const float* in  = (blockIdx.z == 0) ? Wq  : (blockIdx.z == 1) ? Wk  : Wv;
  __bf16*      out = (blockIdx.z == 0) ? Wqt : (blockIdx.z == 1) ? Wkt : Wvt;
  __shared__ __bf16 tile[32][33];
  int tx = threadIdx.x, ty = threadIdx.y;
  int k0 = blockIdx.x * 32, n0 = blockIdx.y * 32;
#pragma unroll
  for (int i = 0; i < 4; ++i) {
    int k = ty * 4 + i;
    tile[k][tx] = (__bf16)in[(k0 + k) * DMODEL + n0 + tx];
  }
  __syncthreads();
#pragma unroll
  for (int i = 0; i < 4; ++i) {
    int nn = ty * 4 + i;
    out[(n0 + nn) * DMODEL + k0 + tx] = tile[tx][nn];
  }
}

// ---------------------------------------------------------------------------
// Kernel 3: QKV projection GEMM.  C(4096x1024) = Xb(bf16) @ W + bias.
// 4 waves/block, each wave owns a 64x64 tile (4x4 WMMA accumulators).
// z=0 -> Q [b,h,s,d], z=1 -> K [b,h,s,d], z=2 -> V transposed [b,h,d,s]
// (transposed V store packs 8 consecutive s into one b128 store).
// ---------------------------------------------------------------------------
__global__ __launch_bounds__(128) void qkv_gemm(
    const __bf16* __restrict__ Xb,
    const __bf16* __restrict__ Wqt, const __bf16* __restrict__ Wkt,
    const __bf16* __restrict__ Wvt,
    const float* __restrict__ bq, const float* __restrict__ bk,
    const float* __restrict__ bv,
    __bf16* __restrict__ Qb, __bf16* __restrict__ Kb,
    __bf16* __restrict__ Vtb) {
  const int z = blockIdx.z;
  const __bf16* Wt   = (z == 0) ? Wqt : (z == 1) ? Wkt : Wvt;
  const float*  bias = (z == 0) ? bq  : (z == 1) ? bk  : bv;

  const int lane = threadIdx.x & 31;
  const int wid  = threadIdx.x >> 5;
  const int ncol = lane & 15;
  const int lh   = lane >> 4;  // lane half
  const int m0 = blockIdx.x * 128 + (wid >> 1) * 64;
  const int n0 = blockIdx.y * 128 + (wid & 1) * 64;

  v8f acc[4][4];
#pragma unroll
  for (int mi = 0; mi < 4; ++mi)
#pragma unroll
    for (int ni = 0; ni < 4; ++ni) acc[mi][ni] = v8f{};

  for (int k0 = 0; k0 < DMODEL; k0 += 32) {
    v16bf afr[4], bfr[4];
#pragma unroll
    for (int mi = 0; mi < 4; ++mi)
      afr[mi] = ldA(Xb + (m0 + mi * 16 + ncol) * DMODEL + k0 + lh * 16);
#pragma unroll
    for (int ni = 0; ni < 4; ++ni)
      bfr[ni] = ldB(Wt + (n0 + ni * 16 + ncol) * DMODEL + k0 + lh * 16);
#pragma unroll
    for (int mi = 0; mi < 4; ++mi)
#pragma unroll
      for (int ni = 0; ni < 4; ++ni)
        acc[mi][ni] = wmma_bf16(afr[mi], bfr[ni], acc[mi][ni]);
  }

  float bn[4];
#pragma unroll
  for (int ni = 0; ni < 4; ++ni) bn[ni] = bias[n0 + ni * 16 + ncol];

  if (z == 2) {  // V, store transposed: Vt[b,h,d,s]
#pragma unroll
    for (int mi = 0; mi < 4; ++mi)
#pragma unroll
      for (int ni = 0; ni < 4; ++ni) {
        int ng = n0 + ni * 16 + ncol;
        int h = ng >> 6, d = ng & 63;
        int m = m0 + mi * 16 + lh * 8;
        int b = m >> 11, s = m & (SEQ - 1);
        v8bf pv;
#pragma unroll
        for (int g = 0; g < 8; ++g) pv[g] = (__bf16)(acc[mi][ni][g] + bn[ni]);
        *(v8bf*)(Vtb + ((size_t)((b * NHEAD + h) * DHEAD + d)) * SEQ + s) = pv;
      }
  } else {  // Q or K: [b,h,s,d]
    __bf16* Out = (z == 0) ? Qb : Kb;
#pragma unroll
    for (int mi = 0; mi < 4; ++mi)
#pragma unroll
      for (int ni = 0; ni < 4; ++ni) {
        int ng = n0 + ni * 16 + ncol;
        int h = ng >> 6, d = ng & 63;
#pragma unroll
        for (int g = 0; g < 8; ++g) {
          int m = m0 + mi * 16 + lh * 8 + g;
          int b = m >> 11, s = m & (SEQ - 1);
          Out[((size_t)((b * NHEAD + h) * SEQ + s)) * DHEAD + d] =
              (__bf16)(acc[mi][ni][g] + bn[ni]);
        }
      }
  }
}

// ---------------------------------------------------------------------------
// Kernel 4: flash attention (causal).  Grid (S/64, B*H), 4 waves/block.
// Each wave: 16 query rows x full head.  Key chunks of 32:
//   scores 16x32 via 4 WMMA, online softmax with shfl row-reductions,
//   P transposed through wave-private LDS, P@V via 4 WMMA into fp32 O.
// ---------------------------------------------------------------------------
__global__ __launch_bounds__(128) void flash_attn(
    const __bf16* __restrict__ Qb, const __bf16* __restrict__ Kb,
    const __bf16* __restrict__ Vtb, float* __restrict__ out) {
  __shared__ __align__(16) __bf16 pbuf[4 * 16 * LDS_STRIDE];

  const int bh   = blockIdx.y;
  const int lane = threadIdx.x & 31;
  const int wid  = threadIdx.x >> 5;
  const int ncol = lane & 15;
  const int lh   = lane >> 4;
  const int q0   = blockIdx.x * 64 + wid * 16;

  const __bf16* Qh = Qb  + (size_t)bh * SEQ * DHEAD;
  const __bf16* Kh = Kb  + (size_t)bh * SEQ * DHEAD;
  const __bf16* Vh = Vtb + (size_t)bh * DHEAD * SEQ;

  v16bf aq[2];
#pragma unroll
  for (int f = 0; f < 2; ++f)
    aq[f] = ldA(Qh + (q0 + ncol) * DHEAD + f * 32 + lh * 16);

  v8f o[4];
#pragma unroll
  for (int t = 0; t < 4; ++t) o[t] = v8f{};
  float mrow[8], lrow[8];
#pragma unroll
  for (int g = 0; g < 8; ++g) { mrow[g] = -1.0e30f; lrow[g] = 0.0f; }

  __bf16* pw = pbuf + wid * (16 * LDS_STRIDE);

  for (int kc = 0; kc <= q0 + 15; kc += 32) {
    if (kc + 32 <= q0 + 15) {  // prefetch next K chunk
      __builtin_prefetch(Kh + (kc + 32 + ncol) * DHEAD, 0, 1);
      __builtin_prefetch(Kh + (kc + 48 + ncol) * DHEAD, 0, 1);
    }
    // ---- scores: S = Q @ K^T, tile 16(q) x 32(keys) ----
    v8f s0 = v8f{}, s1 = v8f{};
#pragma unroll
    for (int f = 0; f < 2; ++f) {
      v16bf bk0 = ldB(Kh + (kc + ncol)      * DHEAD + f * 32 + lh * 16);
      v16bf bk1 = ldB(Kh + (kc + 16 + ncol) * DHEAD + f * 32 + lh * 16);
      s0 = wmma_bf16(aq[f], bk0, s0);
      s1 = wmma_bf16(aq[f], bk1, s1);
    }
    const float sc = 0.125f;  // 1/sqrt(64)
#pragma unroll
    for (int g = 0; g < 8; ++g) { s0[g] *= sc; s1[g] *= sc; }
    if (kc + 31 > q0) {  // causal mask touches this chunk
#pragma unroll
      for (int g = 0; g < 8; ++g) {
        int q = q0 + lh * 8 + g;
        if (kc + ncol      > q) s0[g] = -1.0e30f;
        if (kc + 16 + ncol > q) s1[g] = -1.0e30f;
      }
    }
    // ---- online softmax (row = lh*8+g, cols across the 16-lane half) ----
#pragma unroll
    for (int g = 0; g < 8; ++g) {
      float rmax = redmax16(fmaxf(s0[g], s1[g]));
      float mn = fmaxf(mrow[g], rmax);
      float corr = __expf(mrow[g] - mn);
      mrow[g] = mn;
      s0[g] = __expf(s0[g] - mn);
      s1[g] = __expf(s1[g] - mn);
      float rs = redsum16(s0[g] + s1[g]);
      lrow[g] = lrow[g] * corr + rs;
#pragma unroll
      for (int t = 0; t < 4; ++t) o[t][g] *= corr;
    }
    // ---- transpose P (C-layout -> A-layout) through wave-private LDS ----
#pragma unroll
    for (int g = 0; g < 8; ++g) {
      int r = lh * 8 + g;
      pw[r * LDS_STRIDE + ncol]      = (__bf16)s0[g];
      pw[r * LDS_STRIDE + 16 + ncol] = (__bf16)s1[g];
    }
    v16bf pa = ldA(pw + ncol * LDS_STRIDE + lh * 16);  // same-wave LDS: in order
    // ---- O += P @ V (V pre-transposed: Vt[d][s]) ----
#pragma unroll
    for (int t = 0; t < 4; ++t) {
      v16bf bvf = ldB(Vh + (t * 16 + ncol) * SEQ + kc + lh * 16);
      o[t] = wmma_bf16(pa, bvf, o[t]);
    }
  }

  // ---- epilogue: O / l -> out[b, q, h*64 + d]  (fp32) ----
  const int b = bh >> 4, h = bh & 15;
#pragma unroll
  for (int t = 0; t < 4; ++t)
#pragma unroll
    for (int g = 0; g < 8; ++g) {
      int q = q0 + lh * 8 + g;
      int d = t * 16 + ncol;
      out[((size_t)(b * SEQ + q)) * DMODEL + h * DHEAD + d] = o[t][g] / lrow[g];
    }
}

// ---------------------------------------------------------------------------
// Host launch
// ---------------------------------------------------------------------------
extern "C" void kernel_launch(void* const* d_in, const int* in_sizes, int n_in,
                              void* d_out, int out_size, void* d_ws,
                              size_t ws_size, hipStream_t stream) {
  (void)in_sizes; (void)n_in; (void)out_size; (void)ws_size;
  const float* x  = (const float*)d_in[0];
  const float* Wq = (const float*)d_in[1];
  const float* bq = (const float*)d_in[2];
  const float* Wk = (const float*)d_in[3];
  const float* bk = (const float*)d_in[4];
  const float* Wv = (const float*)d_in[5];
  const float* bv = (const float*)d_in[6];
  float* out = (float*)d_out;

  char* ws = (char*)d_ws;
  size_t off = 0;
  auto carve = [&](size_t bytes) { void* p = ws + off; off += bytes; return p; };
  const size_t xelems = (size_t)NBATCH * SEQ * DMODEL;  // 4M
  __bf16* Xb  = (__bf16*)carve(xelems * 2);                      // 8 MB
  __bf16* Wqt = (__bf16*)carve((size_t)DMODEL * DMODEL * 2);     // 2 MB
  __bf16* Wkt = (__bf16*)carve((size_t)DMODEL * DMODEL * 2);     // 2 MB
  __bf16* Wvt = (__bf16*)carve((size_t)DMODEL * DMODEL * 2);     // 2 MB
  __bf16* Qb  = (__bf16*)carve(xelems * 2);                      // 8 MB
  __bf16* Kb  = (__bf16*)carve(xelems * 2);                      // 8 MB
  __bf16* Vtb = (__bf16*)carve(xelems * 2);                      // 8 MB

  // 1) x -> bf16
  cvt_f32_to_bf16<<<(int)(xelems / 8 / 256), 256, 0, stream>>>(x, Xb,
                                                               (int)xelems);
  // 2) W -> bf16, transposed
  transpose_cvt_w<<<dim3(DMODEL / 32, DMODEL / 32, 3), dim3(32, 8), 0,
                    stream>>>(Wq, Wk, Wv, Wqt, Wkt, Wvt);
  // 3) Q/K/V projections (z picks matrix; V stored transposed)
  qkv_gemm<<<dim3((NBATCH * SEQ) / 128, DMODEL / 128, 3), 128, 0, stream>>>(
      Xb, Wqt, Wkt, Wvt, bq, bk, bv, Qb, Kb, Vtb);
  // 4) causal flash attention
  flash_attn<<<dim3(SEQ / 64, NBATCH * NHEAD), 128, 0, stream>>>(Qb, Kb, Vtb,
                                                                 out);
}
